// RNNVAEODE_45603962749825
// MI455X (gfx1250) — matrix operand order repaired and checked by reading, not compiled, not run on hardware
//
#include <hip/hip_runtime.h>
#include <hip/hip_bf16.h>

typedef __bf16 bf16_t;
typedef __attribute__((ext_vector_type(16))) __bf16 v16bf;
typedef __attribute__((ext_vector_type(8)))  __bf16 v8bf;
typedef __attribute__((ext_vector_type(8)))  float  v8f;

#define DEVINL __device__ __forceinline__

// ---------------- problem dims ----------------
constexpr int Bsz = 256, TP = 32, TFs = 256;
constexpr int H = 100, HF = 275, HG = 75, HO = 75;
constexpr int NSUBc = 2;
constexpr float SLOPEc = 1.0f / 5.5f;

// ---------------- padded dims -----------------
constexpr int KX  = 128;   // x=[t|y]: 101 -> 128 (K for field L1)
constexpr int NF  = 288;   // HF 275 -> 288 (18 n-tiles; also K for L2/L3)
constexpr int NH  = 112;   // H 100 -> 112 (field output stride, f32)
constexpr int KH  = 128;   // H as K: 100 -> 128
constexpr int NG3 = 304;   // 3H 300 -> 304 (19 n-tiles)
constexpr int SG  = 96;    // HG/HO pad as K / LDS stride
constexpr int TILE = 512;  // bf16 elements per packed 32x16 B tile

// tile counts (KT = K/32 tiles, NT = N/16 tiles)
constexpr int KT_F1=4, NT_F1=18;
constexpr int KT_F2=9, NT_F2=18;
constexpr int KT_F3=9, NT_F3=7;
constexpr int KT_HH=4, NT_HH=19;
constexpr int KT_G1=4, NT_G1=5;
constexpr int KT_G2=3, NT_G2=5;
constexpr int KT_G3=3, NT_G3=1;
constexpr int KT_O1=4, NT_O1=5;
constexpr int KT_O2=3, NT_O2=5;
constexpr int KT_O3=3, NT_O3=1;

constexpr int SZF1 = KT_F1*NT_F1*TILE;    // 36864 bf16
constexpr int SZF2 = KT_F2*NT_F2*TILE;    // 82944
constexpr int SZF3 = KT_F3*NT_F3*TILE;    // 32256
constexpr int SZHH = KT_HH*NT_HH*TILE;    // 38912

// packed-weight region offsets (bf16 elements)
constexpr int OFF_F1 = 0;
constexpr int OFF_F2 = OFF_F1 + SZF1;
constexpr int OFF_F3 = OFF_F2 + SZF2;
constexpr int OFF_HH = OFF_F3 + SZF3;
constexpr int OFF_G1 = OFF_HH + SZHH;
constexpr int OFF_G2 = OFF_G1 + KT_G1*NT_G1*TILE;
constexpr int OFF_G3 = OFF_G2 + KT_G2*NT_G2*TILE;
constexpr int OFF_O1 = OFF_G3 + KT_G3*NT_G3*TILE;
constexpr int OFF_O2 = OFF_O1 + KT_O1*NT_O1*TILE;
constexpr int OFF_O3 = OFF_O2 + KT_O2*NT_O2*TILE;
constexpr int PACK_TOTAL = OFF_O3 + KT_O3*NT_O3*TILE;   // 229888 bf16

constexpr size_t GX_OFF   = (size_t)PACK_TOTAL * 2;                         // f32 gx (B x 2)
constexpr size_t TRAJ_OFF = ((GX_OFF + (size_t)2*Bsz*4 + 255) / 256) * 256; // bf16 traj

// dynamic-LDS byte sizes
constexpr size_t ENC_LDS =
    (size_t)(SZF1 + SZF3 + SZHH + 16*KX + 2*16*NF) * 2 +
    (size_t)(16*NH + 3*16*NH + NF + NF + NH + NG3) * 4;      // ~271 KB
constexpr size_t DEC_LDS =
    (size_t)(SZF1 + SZF3 + 16*KX + 2*16*NF) * 2 +
    (size_t)(16*NH + 3*16*NH + NF + NF + NH) * 4;            // ~192 KB

// round-to-nearest-even f32 -> bf16
DEVINL bf16_t f2bf(float f) {
  unsigned u = __builtin_bit_cast(unsigned, f);
  u += 0x7fffu + ((u >> 16) & 1u);
  unsigned short h = (unsigned short)(u >> 16);
  return __builtin_bit_cast(bf16_t, h);
}

// gfx1250 async global->LDS copy (ASYNCcnt-tracked, per ISA 08_async_tensor).
// ldst is a generic pointer into LDS; its low 32 bits are the LDS byte address.
DEVINL void async_g2l(const void* gsrc, void* ldst, int bytes) {
  const int tid = threadIdx.x;
  for (int off = tid * 16; off < bytes; off += 256 * 16) {
    unsigned lo = (unsigned)(uintptr_t)ldst + (unsigned)off;
    unsigned long long ga = (unsigned long long)(uintptr_t)gsrc + (unsigned)off;
    asm volatile("global_load_async_to_lds_b128 %0, %1, off"
                 :: "v"(lo), "v"(ga) : "memory");
  }
}
DEVINL void async_wait0() {
  asm volatile("s_wait_asynccnt 0x0" ::: "memory");
}

// One 16x16 C tile of A(16xK) * B(Kx16).
// A: bf16 buffer, row stride `rs`. Lane layout per ISA 16-bit A 16x32:
//   m=L&15; halves 0..7 -> K kt*32 + (L>=16?8:0) + 0..7, halves 8..15 -> +16.
// B: packed tiles, lane L reads 16 contiguous bf16 at tile + L*16.
template <int KT>
DEVINL v8f mma_tile(const bf16_t* __restrict__ abuf, int rs,
                    const bf16_t* __restrict__ btiles) {
  const int lane = threadIdx.x & 31;
  const int m    = lane & 15;
  const int koff = (lane >> 4) * 8;
  v8f c = {};
#pragma unroll
  for (int kt = 0; kt < KT; ++kt) {
    const bf16_t* ap = abuf + m * rs + kt * 32 + koff;
    v8bf lo = *(const v8bf*)(ap);
    v8bf hi = *(const v8bf*)(ap + 16);
    v16bf a = __builtin_shufflevector(lo, hi, 0,1,2,3,4,5,6,7,8,9,10,11,12,13,14,15);
    v16bf b = *(const v16bf*)(btiles + kt * TILE + lane * 16);
    c = __builtin_amdgcn_wmma_f32_16x16x32_bf16(false, a, false, b, (short)0, c,
                                                false, false);
  }
  return c;
}

struct FW {
  const bf16_t *w1, *w2, *w3;   // w1/w3 in LDS, w2 in global (L2-resident)
  const float  *b1, *b2, *b3;   // zero-padded bias vectors in LDS
};

// field: k = tanh(W3 tanh(W2 tanh(W1 [t|y] + b1) + b2) + b3)  (16-row batch tile)
DEVINL void field_eval(const FW& fw, bf16_t* sx, bf16_t* sa1, bf16_t* sa2,
                       const float* tvec, const float* yin, float* kout) {
  const int tid  = threadIdx.x;
  const int lane = tid & 31;
  const int wave = tid >> 5;
  const int mb   = (lane >> 4) * 8;

  for (int i = tid; i < 16 * KX; i += 256) {          // x = [t | y | 0]
    int m = i >> 7, cx = i & 127;
    float v = (cx == 0) ? tvec[m] : (cx <= H ? yin[m * NH + (cx - 1)] : 0.f);
    sx[i] = f2bf(v);
  }
  __syncthreads();
  for (int nt = wave; nt < NT_F1; nt += 8) {          // L1 (weights in LDS)
    v8f c = mma_tile<KT_F1>(sx, KX, fw.w1 + nt * KT_F1 * TILE);
    int col = nt * 16 + (lane & 15);
    float bias = fw.b1[col];
#pragma unroll
    for (int r = 0; r < 8; ++r)
      sa1[(mb + r) * NF + col] = f2bf(tanhf(c[r] + bias));
  }
  __syncthreads();
  for (int nt = wave; nt < NT_F2; nt += 8) {          // L2 (weights from L2$)
    v8f c = mma_tile<KT_F2>(sa1, NF, fw.w2 + nt * KT_F2 * TILE);
    int col = nt * 16 + (lane & 15);
    float bias = fw.b2[col];
#pragma unroll
    for (int r = 0; r < 8; ++r)
      sa2[(mb + r) * NF + col] = f2bf(tanhf(c[r] + bias));
  }
  __syncthreads();
  for (int nt = wave; nt < NT_F3; nt += 8) {          // L3 (weights in LDS)
    v8f c = mma_tile<KT_F3>(sa2, NF, fw.w3 + nt * KT_F3 * TILE);
    int col = nt * 16 + (lane & 15);
    float bias = fw.b3[col];
#pragma unroll
    for (int r = 0; r < 8; ++r)
      kout[(mb + r) * NH + col] = tanhf(c[r] + bias);
  }
  __syncthreads();
}

// RK4 with NSUB substeps; y in-place (16 x NH f32 LDS).
DEVINL void rk4_step(const FW& fw, bf16_t* sx, bf16_t* sa1, bf16_t* sa2,
                     float* sy, float* syt, float* sk, float* sacc,
                     const float* st0, const float* sdt, float* stc) {
  const int tid = threadIdx.x;
  for (int sub = 0; sub < NSUBc; ++sub) {
    if (tid < 16) stc[tid] = st0[tid] + (float)sub * sdt[tid];
    __syncthreads();
    field_eval(fw, sx, sa1, sa2, stc, sy, sk);                       // k1
    for (int i = tid; i < 16 * NH; i += 256) {
      int m = i / NH; float dt = sdt[m], k = sk[i];
      sacc[i] = k; syt[i] = sy[i] + 0.5f * dt * k;
    }
    if (tid < 16) stc[tid] = st0[tid] + ((float)sub + 0.5f) * sdt[tid];
    __syncthreads();
    field_eval(fw, sx, sa1, sa2, stc, syt, sk);                      // k2
    for (int i = tid; i < 16 * NH; i += 256) {
      int m = i / NH; float dt = sdt[m], k = sk[i];
      sacc[i] += 2.f * k; syt[i] = sy[i] + 0.5f * dt * k;
    }
    __syncthreads();
    field_eval(fw, sx, sa1, sa2, stc, syt, sk);                      // k3
    for (int i = tid; i < 16 * NH; i += 256) {
      int m = i / NH; float dt = sdt[m], k = sk[i];
      sacc[i] += 2.f * k; syt[i] = sy[i] + dt * k;
    }
    if (tid < 16) stc[tid] = st0[tid] + ((float)sub + 1.f) * sdt[tid];
    __syncthreads();
    field_eval(fw, sx, sa1, sa2, stc, syt, sk);                      // k4
    for (int i = tid; i < 16 * NH; i += 256) {
      int m = i / NH;
      sy[i] += sdt[m] * (sacc[i] + sk[i]) * (1.f / 6.f);
    }
    __syncthreads();
  }
}

// ---------------- weight repack: f32 (Nout x K) -> bf16 WMMA B-tiles ----------------
struct PackDesc { const float* W; int Nr, Kr, KT, NT, off; };
struct PackArgs { PackDesc d[10]; };

__global__ __launch_bounds__(256) void pack_weights(PackArgs pa, bf16_t* __restrict__ pw) {
  for (int e = blockIdx.x * 256 + threadIdx.x; e < PACK_TOTAL; e += gridDim.x * 256) {
    int r = 0;
#pragma unroll
    for (int i = 1; i < 10; ++i) r = (e >= pa.d[i].off) ? i : r;
    PackDesc D = pa.d[r];
    int le = e - D.off;
    int tile = le >> 9;
    int L = (le >> 4) & 31;
    int j = le & 15;
    int nt = tile / D.KT;
    int kt = tile - nt * D.KT;
    int n = nt * 16 + (L & 15);
    int k = kt * 32 + ((L >> 4) << 3) + (j & 7) + ((j >> 3) << 4);   // lanes 0-15 hold K 0..7,16..23; lanes 16-31 K 8..15,24..31
    float v = (n < D.Nr && k < D.Kr) ? D.W[n * D.Kr + k] : 0.f;
    pw[e] = f2bf(v);
  }
}

// ---------------- encoder: 32x (RK4 + GRU), then generative head -> gx (B x 2) --------
__global__ __launch_bounds__(256) void ode_encoder(
    const float* __restrict__ past, const float* __restrict__ h0,
    const bf16_t* __restrict__ pw,
    const float* fb1, const float* fb2, const float* fb3,
    const float* Wih, const float* bih, const float* bhh,
    const float* gb1, const float* gb2, const float* gb3,
    float* __restrict__ gxbuf) {
  extern __shared__ char dynlds[];
  bf16_t* lF1 = (bf16_t*)dynlds;
  bf16_t* lF3 = lF1 + SZF1;
  bf16_t* lHH = lF3 + SZF3;
  bf16_t* sx  = lHH + SZHH;
  bf16_t* sa1 = sx  + 16 * KX;
  bf16_t* sa2 = sa1 + 16 * NF;
  float*  sy  = (float*)(sa2 + 16 * NF);
  float*  sf  = sy + 16 * NH;            // yt | k | acc ; reused as GRU gh
  float*  lb1 = sf + 3 * 16 * NH;
  float*  lb2 = lb1 + NF;
  float*  lb3 = lb2 + NF;
  float*  lbh = lb3 + NH;
  __shared__ float st0[16], sdt[16], stc[16], sxd[16];

  const int tid = threadIdx.x, wg = blockIdx.x;
  const int lane = tid & 31, wave = tid >> 5, mb = (lane >> 4) * 8;
  float* syt = sf; float* sk = sf + 16 * NH; float* sacc = sf + 2 * 16 * NH;
  float* sgh = sf;

  // stage hot weights to LDS via gfx1250 async loads (ASYNCcnt)
  async_g2l(pw + OFF_F1, lF1, SZF1 * 2);
  async_g2l(pw + OFF_F3, lF3, SZF3 * 2);
  async_g2l(pw + OFF_HH, lHH, SZHH * 2);
  // zero-padded bias vectors + initial state while async copies are in flight
  for (int i = tid; i < NF; i += 256) {
    lb1[i] = (i < HF) ? fb1[i] : 0.f;
    lb2[i] = (i < HF) ? fb2[i] : 0.f;
  }
  for (int i = tid; i < NH; i += 256)  lb3[i] = (i < H) ? fb3[i] : 0.f;
  for (int i = tid; i < NG3; i += 256) lbh[i] = (i < 3 * H) ? bhh[i] : 0.f;
  for (int i = tid; i < 16 * NH; i += 256) {
    int m = i / NH, cx = i % NH;
    sy[i] = (cx < H) ? h0[(wg * 16 + m) * H + cx] : 0.f;
  }
  // keep big F2 hot in cache
  for (int off = tid * 128; off < SZF2 * 2; off += 256 * 128)
    __builtin_prefetch((const char*)(pw + OFF_F2) + off, 0, 3);
  async_wait0();
  __syncthreads();

  FW fw{lF1, pw + OFF_F2, lF3, lb1, lb2, lb3};

  for (int t = 0; t < TP; ++t) {
    if (tid < 16) {
      int b = wg * 16 + tid;
      float t1 = past[(b * TP + t) * 2 + 0];
      float t0 = (t == 0) ? (t1 - 1.f) : past[(b * TP + t - 1) * 2 + 0];
      st0[tid] = t0;
      sdt[tid] = (t1 - t0) * (1.f / NSUBc);
      sxd[tid] = past[(b * TP + t) * 2 + 1];
    }
    __syncthreads();
    rk4_step(fw, sx, sa1, sa2, sy, syt, sk, sacc, st0, sdt, stc);
    // ---- GRU: gh = h @ Whh.T + bhh (WMMA from LDS weights), gi elementwise (D=1) ----
    for (int i = tid; i < 16 * KH; i += 256) {
      int m = i >> 7, cx = i & 127;
      sx[i] = f2bf(cx < H ? sy[m * NH + cx] : 0.f);
    }
    __syncthreads();
    for (int nt = wave; nt < NT_HH; nt += 8) {
      v8f c = mma_tile<KT_HH>(sx, KH, lHH + nt * KT_HH * TILE);
      int col = nt * 16 + (lane & 15);
      float bias = lbh[col];
#pragma unroll
      for (int r = 0; r < 8; ++r) sgh[(mb + r) * NG3 + col] = c[r] + bias;
    }
    __syncthreads();
    for (int i = tid; i < 16 * H; i += 256) {
      int m = i / H, j = i % H;
      float xv = sxd[m];
      float gir = xv * Wih[j]         + bih[j];
      float giz = xv * Wih[H + j]     + bih[H + j];
      float gin = xv * Wih[2 * H + j] + bih[2 * H + j];
      const float* gh = sgh + m * NG3;
      float rr = 1.f / (1.f + expf(-(gir + gh[j])));
      float zz = 1.f / (1.f + expf(-(giz + gh[H + j])));
      float nn = tanhf(gin + rr * gh[2 * H + j]);
      float hv = sy[m * NH + j];
      sy[m * NH + j] = (1.f - zz) * nn + zz * hv;
    }
    __syncthreads();
  }

  // ---- generative head: leaky MLP 100 -> 75 -> 75 -> 2 (weights from global) ----
  for (int i = tid; i < 16 * SG; i += 256) { sa1[i] = f2bf(0.f); sa2[i] = f2bf(0.f); }
  for (int i = tid; i < 16 * KH; i += 256) {
    int m = i >> 7, cx = i & 127;
    sx[i] = f2bf(cx < H ? sy[m * NH + cx] : 0.f);
  }
  __syncthreads();
  for (int nt = wave; nt < NT_G1; nt += 8) {
    v8f c = mma_tile<KT_G1>(sx, KH, pw + OFF_G1 + nt * KT_G1 * TILE);
    int col = nt * 16 + (lane & 15);
    float bias = (col < HG) ? gb1[col] : 0.f;
#pragma unroll
    for (int r = 0; r < 8; ++r) {
      float v = c[r] + bias; v = (v >= 0.f) ? v : SLOPEc * v;
      sa1[(mb + r) * SG + col] = f2bf(v);
    }
  }
  __syncthreads();
  for (int nt = wave; nt < NT_G2; nt += 8) {
    v8f c = mma_tile<KT_G2>(sa1, SG, pw + OFF_G2 + nt * KT_G2 * TILE);
    int col = nt * 16 + (lane & 15);
    float bias = (col < HG) ? gb2[col] : 0.f;
#pragma unroll
    for (int r = 0; r < 8; ++r) {
      float v = c[r] + bias; v = (v >= 0.f) ? v : SLOPEc * v;
      sa2[(mb + r) * SG + col] = f2bf(v);
    }
  }
  __syncthreads();
  if (wave == 0) {
    v8f c = mma_tile<KT_G3>(sa2, SG, pw + OFF_G3);
    int col = lane & 15;
    if (col < 2) {
      float bias = gb3[col];
#pragma unroll
      for (int r = 0; r < 8; ++r)
        gxbuf[(wg * 16 + mb + r) * 2 + col] = c[r] + bias;
    }
  }
}

// ---------------- decoder: z0 from gx interleave + eps, 255x RK4, traj(bf16) -----------
__global__ __launch_bounds__(256) void ode_decoder(
    const float* __restrict__ t_future, const float* __restrict__ eps,
    const float* __restrict__ gxbuf, const bf16_t* __restrict__ pw,
    const float* fb1, const float* fb2, const float* fb3,
    bf16_t* __restrict__ traj) {
  extern __shared__ char dynlds[];
  bf16_t* lF1 = (bf16_t*)dynlds;
  bf16_t* lF3 = lF1 + SZF1;
  bf16_t* sx  = lF3 + SZF3;
  bf16_t* sa1 = sx  + 16 * KX;
  bf16_t* sa2 = sa1 + 16 * NF;
  float*  sy  = (float*)(sa2 + 16 * NF);
  float*  sf  = sy + 16 * NH;
  float*  lb1 = sf + 3 * 16 * NH;
  float*  lb2 = lb1 + NF;
  float*  lb3 = lb2 + NF;
  __shared__ float st0[16], sdt[16], stc[16], sloc[16], sscl[16];

  const int tid = threadIdx.x, wg = blockIdx.x;
  float* syt = sf; float* sk = sf + 16 * NH; float* sacc = sf + 2 * 16 * NH;

  async_g2l(pw + OFF_F1, lF1, SZF1 * 2);
  async_g2l(pw + OFF_F3, lF3, SZF3 * 2);
  for (int i = tid; i < NF; i += 256) {
    lb1[i] = (i < HF) ? fb1[i] : 0.f;
    lb2[i] = (i < HF) ? fb2[i] : 0.f;
  }
  for (int i = tid; i < NH; i += 256) lb3[i] = (i < H) ? fb3[i] : 0.f;
  for (int off = tid * 128; off < SZF2 * 2; off += 256 * 128)
    __builtin_prefetch((const char*)(pw + OFF_F2) + off, 0, 3);

  if (tid < 16) {   // reference: pr = concat(gx[:,0], |gx[:,1]|).reshape(-1,2)
    int b = wg * 16 + tid;
    int i0 = 2 * b, i1 = 2 * b + 1;
    sloc[tid] = (i0 < Bsz) ? gxbuf[i0 * 2 + 0] : fabsf(gxbuf[(i0 - Bsz) * 2 + 1]);
    sscl[tid] = (i1 < Bsz) ? gxbuf[i1 * 2 + 0] : fabsf(gxbuf[(i1 - Bsz) * 2 + 1]);
  }
  async_wait0();
  __syncthreads();

  FW fw{lF1, pw + OFF_F2, lF3, lb1, lb2, lb3};

  for (int i = tid; i < 16 * NH; i += 256) {
    int m = i / NH, cx = i % NH;
    int b = wg * 16 + m;
    sy[i] = (cx < H) ? sloc[m] + sscl[m] * eps[cx * Bsz + b] : 0.f;
  }
  __syncthreads();
  for (int i = tid; i < 16 * 128; i += 256) {
    int m = i >> 7, cx = i & 127;
    traj[((size_t)(wg * 16 + m) * TFs + 0) * 128 + cx] =
        f2bf(cx < H ? sy[m * NH + cx] : 0.f);
  }
  __syncthreads();
  for (int s = 1; s < TFs; ++s) {
    if (tid < 16) {
      int b = wg * 16 + tid;
      float t0 = t_future[b * TFs + s - 1], t1 = t_future[b * TFs + s];
      st0[tid] = t0; sdt[tid] = (t1 - t0) * (1.f / NSUBc);
    }
    __syncthreads();
    rk4_step(fw, sx, sa1, sa2, sy, syt, sk, sacc, st0, sdt, stc);
    for (int i = tid; i < 16 * 128; i += 256) {
      int m = i >> 7, cx = i & 127;
      traj[((size_t)(wg * 16 + m) * TFs + s) * 128 + cx] =
          f2bf(cx < H ? sy[m * NH + cx] : 0.f);
    }
    __syncthreads();
  }
}

// ---------------- output MLP: (B*TF x 100) -> 75 -> 75 -> 1, one M-tile per wave -------
__global__ __launch_bounds__(256) void out_mlp(
    const bf16_t* __restrict__ traj, const bf16_t* __restrict__ pw,
    const float* ob1, const float* ob2, const float* ob3,
    float* __restrict__ out) {
  __shared__ bf16_t so1[8][16 * SG];
  __shared__ bf16_t so2[8][16 * SG];
  const int tid = threadIdx.x, lane = tid & 31, wave = tid >> 5, mb = (lane >> 4) * 8;
  const int mt = blockIdx.x * 8 + wave;          // 4096 M-tiles total
  bf16_t* o1 = so1[wave]; bf16_t* o2 = so2[wave];
  for (int i = lane; i < 16 * SG; i += 32) { o1[i] = f2bf(0.f); o2[i] = f2bf(0.f); }
  const bf16_t* A = traj + (size_t)mt * 16 * 128;
  __builtin_prefetch(A, 0, 0);                   // gfx1250 global_prefetch_b8
#pragma unroll
  for (int nt = 0; nt < NT_O1; ++nt) {
    v8f c = mma_tile<KT_O1>(A, 128, pw + OFF_O1 + nt * KT_O1 * TILE);
    int col = nt * 16 + (lane & 15);
    float bias = (col < HO) ? ob1[col] : 0.f;
#pragma unroll
    for (int r = 0; r < 8; ++r) {
      float v = c[r] + bias; v = (v >= 0.f) ? v : SLOPEc * v;
      o1[(mb + r) * SG + col] = f2bf(v);
    }
  }
#pragma unroll
  for (int nt = 0; nt < NT_O2; ++nt) {
    v8f c = mma_tile<KT_O2>(o1, SG, pw + OFF_O2 + nt * KT_O2 * TILE);
    int col = nt * 16 + (lane & 15);
    float bias = (col < HO) ? ob2[col] : 0.f;
#pragma unroll
    for (int r = 0; r < 8; ++r) {
      float v = c[r] + bias; v = (v >= 0.f) ? v : SLOPEc * v;
      o2[(mb + r) * SG + col] = f2bf(v);
    }
  }
  {
    v8f c = mma_tile<KT_O3>(o2, SG, pw + OFF_O3);
    if ((lane & 15) == 0) {
      float bias = ob3[0];
#pragma unroll
      for (int r = 0; r < 8; ++r)
        out[mt * 16 + mb + r] = c[r] + bias;
    }
  }
}

extern "C" void kernel_launch(void* const* d_in, const int* in_sizes, int n_in,
                              void* d_out, int out_size, void* d_ws, size_t ws_size,
                              hipStream_t stream) {
  (void)in_sizes; (void)n_in; (void)out_size; (void)ws_size;
  const float* past     = (const float*)d_in[0];
  const float* h0       = (const float*)d_in[1];
  const float* t_future = (const float*)d_in[2];
  const float* eps      = (const float*)d_in[3];
  const float* fW1 = (const float*)d_in[4];  const float* fb1 = (const float*)d_in[5];
  const float* fW2 = (const float*)d_in[6];  const float* fb2 = (const float*)d_in[7];
  const float* fW3 = (const float*)d_in[8];  const float* fb3 = (const float*)d_in[9];
  const float* Wih = (const float*)d_in[10]; const float* Whh = (const float*)d_in[11];
  const float* bih = (const float*)d_in[12]; const float* bhh = (const float*)d_in[13];
  const float* gW1 = (const float*)d_in[14]; const float* gb1 = (const float*)d_in[15];
  const float* gW2 = (const float*)d_in[16]; const float* gb2 = (const float*)d_in[17];
  const float* gW3 = (const float*)d_in[18]; const float* gb3 = (const float*)d_in[19];
  const float* oW1 = (const float*)d_in[20]; const float* ob1 = (const float*)d_in[21];
  const float* oW2 = (const float*)d_in[22]; const float* ob2 = (const float*)d_in[23];
  const float* oW3 = (const float*)d_in[24]; const float* ob3 = (const float*)d_in[25];

  bf16_t* pw    = (bf16_t*)d_ws;
  float*  gxbuf = (float*)((char*)d_ws + GX_OFF);
  bf16_t* traj  = (bf16_t*)((char*)d_ws + TRAJ_OFF);

  PackArgs pa;
  pa.d[0] = {fW1, HF,  H + 1, KT_F1, NT_F1, OFF_F1};
  pa.d[1] = {fW2, HF,  HF,    KT_F2, NT_F2, OFF_F2};
  pa.d[2] = {fW3, H,   HF,    KT_F3, NT_F3, OFF_F3};
  pa.d[3] = {Whh, 3*H, H,     KT_HH, NT_HH, OFF_HH};
  pa.d[4] = {gW1, HG,  H,     KT_G1, NT_G1, OFF_G1};
  pa.d[5] = {gW2, HG,  HG,    KT_G2, NT_G2, OFF_G2};
  pa.d[6] = {gW3, 2,   HG,    KT_G3, NT_G3, OFF_G3};
  pa.d[7] = {oW1, HO,  H,     KT_O1, NT_O1, OFF_O1};
  pa.d[8] = {oW2, HO,  HO,    KT_O2, NT_O2, OFF_O2};
  pa.d[9] = {oW3, 1,   HO,    KT_O3, NT_O3, OFF_O3};

  (void)hipFuncSetAttribute((const void*)ode_encoder,
      hipFuncAttributeMaxDynamicSharedMemorySize, (int)ENC_LDS);
  (void)hipFuncSetAttribute((const void*)ode_decoder,
      hipFuncAttributeMaxDynamicSharedMemorySize, (int)DEC_LDS);

  pack_weights<<<(PACK_TOTAL + 255) / 256, 256, 0, stream>>>(pa, pw);
  ode_encoder<<<Bsz / 16, 256, ENC_LDS, stream>>>(past, h0, pw, fb1, fb2, fb3,
                                                  Wih, bih, bhh, gb1, gb2, gb3, gxbuf);
  ode_decoder<<<Bsz / 16, 256, DEC_LDS, stream>>>(t_future, eps, gxbuf, pw,
                                                  fb1, fb2, fb3, traj);
  out_mlp<<<(Bsz * TFs / 16) / 8, 256, 0, stream>>>(traj, pw, ob1, ob2, ob3,
                                                    (float*)d_out);
}
